// JointDrifting_10050223472719
// MI455X (gfx1250) — matrix operand order repaired and checked by reading, not compiled, not run on hardware
//
#include <hip/hip_runtime.h>
#include <hip/hip_bf16.h>
#include <math.h>

typedef __attribute__((ext_vector_type(16))) __bf16 v16bf;
typedef __attribute__((ext_vector_type(8)))  float  v8f;
typedef __attribute__((ext_vector_type(4)))  unsigned int u32x4;

#define NPTS   4096
#define DDIM   512
#define TWO_N  8192
#define TAU_INV 10.0f
#define SELF_MASK 1000000.0f

#define BM 128
#define BN 128
#define BK 32
#define LDK 40      // halves per padded LDS row: 80 bytes (16B-aligned, 20-bank stride)

// ---------------- CDNA5 async global->LDS copy (ASYNCcnt path) ----------------

__device__ inline unsigned lds_off(const void* p) {
    // generic pointer to LDS: low 32 bits are the LDS byte offset
    return (unsigned)(unsigned long long)(uintptr_t)p;
}

__device__ inline void async_copy_b128(unsigned dst_lds, const void* gsrc) {
    asm volatile("global_load_async_to_lds_b128 %0, %1, off"
                 :: "v"(dst_lds), "v"(gsrc)
                 : "memory");
}

__device__ inline void wait_async0() {
    asm volatile("s_wait_asynccnt 0x0" ::: "memory");
}

// Copy one 128-row x 32-half tile (row stride DDIM in global, LDK in LDS),
// 16B chunks, 2 chunks per thread (256 threads, 512 chunks).
__device__ inline void stage_tile_async(const __bf16* __restrict__ gsrc, int grow0, int kk,
                                        __bf16* ldst, int tid) {
    #pragma unroll
    for (int t = 0; t < 2; ++t) {
        int idx = tid + t * 256;            // 0..511
        int r = idx >> 2, c = idx & 3;      // 4 x 16B chunks per 64B row
        const __bf16* g = gsrc + (size_t)(grow0 + r) * DDIM + kk + c * 8;
        async_copy_b128(lds_off(ldst + r * LDK + c * 8), g);
    }
}

// ---------------- fragment loaders (CDNA5 wave32 WMMA layouts) ----------------

// A-matrix 16x32 bf16: lane L -> M = L%16; lanes 0-15 hold K=0..7,16..23,
// lanes 16-31 hold K=8..15,24..31. Two contiguous 8-half (16B) runs.
__device__ inline v16bf load_a_frag(const __bf16* __restrict__ As, int rowbase, int lane) {
    const __bf16* p = As + (rowbase + (lane & 15)) * LDK + ((lane >> 4) & 1) * 8;
    union { v16bf v; u32x4 q[2]; } u;
    u.q[0] = *(const u32x4*)(p);
    u.q[1] = *(const u32x4*)(p + 16);
    return u.v;
}

// B-matrix 32x16 bf16 from LDS stored transposed as [n][k]:
// lane L -> N = L%16; lanes 0-15 hold K=0..7,16..23, lanes 16-31 hold K=8..15,24..31.
// Two contiguous 8-half (16B) runs -> two 16B loads.
__device__ inline v16bf load_bt_frag(const __bf16* __restrict__ Bs, int colbase, int lane) {
    const __bf16* p = Bs + (colbase + (lane & 15)) * LDK + ((lane >> 4) & 1) * 8;
    union { v16bf v; u32x4 q[2]; } u;
    u.q[0] = *(const u32x4*)(p);
    u.q[1] = *(const u32x4*)(p + 16);
    return u.v;
}

// ---------------- kernel 1: fp32 -> bf16 conversion + row squared norms ----------------

__global__ __launch_bounds__(256) void cvt_norm_kernel(
    const float* __restrict__ x, const float* __restrict__ y,
    __bf16* __restrict__ xbf, __bf16* __restrict__ ybf,
    float* __restrict__ nx, float* __restrict__ ny) {
    int row = blockIdx.x;
    const float* src = (blockIdx.y == 0) ? x : y;
    __bf16*      dst = (blockIdx.y == 0) ? xbf : ybf;
    float*       nrm = (blockIdx.y == 0) ? nx : ny;
    __shared__ float red[256];
    float s = 0.f;
    for (int c = threadIdx.x; c < DDIM; c += 256) {
        float v = src[row * DDIM + c];
        dst[row * DDIM + c] = (__bf16)v;
        s += v * v;
    }
    red[threadIdx.x] = s;
    __syncthreads();
    for (int off = 128; off > 0; off >>= 1) {
        if (threadIdx.x < off) red[threadIdx.x] += red[threadIdx.x + off];
        __syncthreads();
    }
    if (threadIdx.x == 0) nrm[row] = red[0];
}

// ---------------- kernel 2: logits via WMMA GEMM, async double-buffered ----------------

__global__ __launch_bounds__(256) void logits_kernel(
    const __bf16* __restrict__ xbf, const __bf16* __restrict__ ybf,
    const float* __restrict__ nx, const float* __restrict__ ny,
    float* __restrict__ L) {
    __shared__ alignas(16) __bf16 As[2][BM * LDK];
    __shared__ alignas(16) __bf16 Bs[2][BN * LDK];
    int row0 = blockIdx.y * BM;
    int col0 = blockIdx.x * BN;            // global column in [0, 2N)
    bool isNeg = (col0 >= NPTS);
    const __bf16* bsrc = isNeg ? xbf : ybf;
    int bcol0 = isNeg ? (col0 - NPTS) : col0;

    int tid  = threadIdx.x;
    int lane = tid & 31;
    int wave = tid >> 5;
    int wm = (wave & 3) * 32;              // wave row offset in tile
    int wn = (wave >> 2) * 64;             // wave col offset in tile

    v8f acc[2][4];
    #pragma unroll
    for (int mi = 0; mi < 2; ++mi)
        #pragma unroll
        for (int ni = 0; ni < 4; ++ni) { v8f z = {}; acc[mi][ni] = z; }

    // prologue: async-stage first K tile
    stage_tile_async(xbf,  row0,  0, As[0], tid);
    stage_tile_async(bsrc, bcol0, 0, Bs[0], tid);

    int cur = 0;
    for (int kk = 0; kk < DDIM; kk += BK) {
        wait_async0();                     // my async copies landed
        __syncthreads();                   // everyone's copies landed
        int nxt = cur ^ 1;
        if (kk + BK < DDIM) {              // prefetch next tile into other buffer
            stage_tile_async(xbf,  row0,  kk + BK, As[nxt], tid);
            stage_tile_async(bsrc, bcol0, kk + BK, Bs[nxt], tid);
        }
        v16bf af[2], bfr[4];
        #pragma unroll
        for (int mi = 0; mi < 2; ++mi) af[mi]  = load_a_frag(As[cur], wm + mi * 16, lane);
        #pragma unroll
        for (int ni = 0; ni < 4; ++ni) bfr[ni] = load_bt_frag(Bs[cur], wn + ni * 16, lane);
        #pragma unroll
        for (int mi = 0; mi < 2; ++mi)
            #pragma unroll
            for (int ni = 0; ni < 4; ++ni)
                acc[mi][ni] = __builtin_amdgcn_wmma_f32_16x16x32_bf16(
                    false, af[mi], false, bfr[ni], (short)0, acc[mi][ni], false, false);
        __syncthreads();                   // done reading cur before it is overwritten
        cur = nxt;
    }

    int Mb = (lane >> 4) * 8;
    int nl = lane & 15;
    #pragma unroll
    for (int mi = 0; mi < 2; ++mi)
        #pragma unroll
        for (int ni = 0; ni < 4; ++ni)
            #pragma unroll
            for (int r = 0; r < 8; ++r) {
                int i = row0 + wm + mi * 16 + Mb + r;
                int j = col0 + wn + ni * 16 + nl;
                float g  = acc[mi][ni][r];
                float nj = isNeg ? nx[j - NPTS] : ny[j];
                float sq = nx[i] + nj - 2.f * g;
                float d  = sqrtf(fmaxf(sq, 1e-12f));
                if (isNeg && (j - NPTS) == i) d += SELF_MASK;
                L[(size_t)i * TWO_N + j] = -d * TAU_INV;
            }
}

// ---------------- kernels 3/4: online logsumexp over rows / columns ----------------

__device__ inline void lse_update(float v, float& m, float& s) {
    if (v > m) { s = s * __expf(m - v) + 1.f; m = v; }
    else       { s += __expf(v - m); }
}

__global__ __launch_bounds__(256) void row_lse_kernel(
    const float* __restrict__ L, float* __restrict__ rowLSE) {
    int i = blockIdx.x;
    float m = -3.402823466e38f, s = 0.f;
    for (int j = threadIdx.x; j < TWO_N; j += 256)
        lse_update(L[(size_t)i * TWO_N + j], m, s);
    __shared__ float sm[256], ss[256];
    sm[threadIdx.x] = m; ss[threadIdx.x] = s;
    __syncthreads();
    for (int off = 128; off > 0; off >>= 1) {
        if (threadIdx.x < off) {
            float m1 = sm[threadIdx.x], s1 = ss[threadIdx.x];
            float m2 = sm[threadIdx.x + off], s2 = ss[threadIdx.x + off];
            float mm = fmaxf(m1, m2);
            sm[threadIdx.x] = mm;
            ss[threadIdx.x] = s1 * __expf(m1 - mm) + s2 * __expf(m2 - mm);
        }
        __syncthreads();
    }
    if (threadIdx.x == 0) rowLSE[i] = sm[0] + __logf(ss[0]);
}

__global__ __launch_bounds__(256) void col_lse_kernel(
    const float* __restrict__ L, float* __restrict__ colLSE) {
    int j = blockIdx.x * 256 + threadIdx.x;
    float m = -3.402823466e38f, s = 0.f;
    for (int i = 0; i < NPTS; ++i)
        lse_update(L[(size_t)i * TWO_N + j], m, s);
    colLSE[j] = m + __logf(s);
}

// ---------------- kernel 5: drift = A_pos @ y - A_neg @ x (sign folded into A) ----------------

__global__ __launch_bounds__(256) void drift_kernel(
    const __bf16* __restrict__ xbf, const __bf16* __restrict__ ybf,
    const float* __restrict__ L, const float* __restrict__ rowLSE,
    const float* __restrict__ colLSE, float* __restrict__ out) {
    __shared__ alignas(16) __bf16 Ash[BM * LDK];   // 128 rows x 32 k, A = +/- exp(...)
    __shared__ alignas(16) __bf16 Bsh[BN * LDK];   // transposed [n][k]: [y;x] tile
    int row0 = blockIdx.y * BM;
    int d0   = blockIdx.x * BN;

    int tid  = threadIdx.x;
    int lane = tid & 31;
    int wave = tid >> 5;
    int wm = (wave & 3) * 32;
    int wn = (wave >> 2) * 64;

    v8f acc[2][4];
    #pragma unroll
    for (int mi = 0; mi < 2; ++mi)
        #pragma unroll
        for (int ni = 0; ni < 4; ++ni) { v8f z = {}; acc[mi][ni] = z; }

    for (int kk = 0; kk < TWO_N; kk += BK) {
        bool neg = (kk >= NPTS);
        const __bf16* src = neg ? xbf : ybf;
        int   j0   = neg ? (kk - NPTS) : kk;
        float sign = neg ? -1.f : 1.f;
        // A tile: regenerate A = sign * exp(l - 0.5*(LSEr + LSEc)), cast to bf16
        #pragma unroll
        for (int t = 0; t < 16; ++t) {
            int idx = tid + t * 256;
            int r = idx >> 5, k = idx & 31;
            int i = row0 + r, j = kk + k;
            float v = __expf(L[(size_t)i * TWO_N + j] - 0.5f * (rowLSE[i] + colLSE[j]));
            Ash[r * LDK + k] = (__bf16)(sign * v);
        }
        // B tile (transposed store): Bsh[n][k] = src[j0+k][d0+n]; coalesced global reads
        #pragma unroll
        for (int t = 0; t < 16; ++t) {
            int idx = tid + t * 256;
            int n = idx & 127, k = idx >> 7;
            Bsh[n * LDK + k] = src[(size_t)(j0 + k) * DDIM + d0 + n];
        }
        __syncthreads();
        v16bf af[2], bfr[4];
        #pragma unroll
        for (int mi = 0; mi < 2; ++mi) af[mi]  = load_a_frag(Ash, wm + mi * 16, lane);
        #pragma unroll
        for (int ni = 0; ni < 4; ++ni) bfr[ni] = load_bt_frag(Bsh, wn + ni * 16, lane);
        #pragma unroll
        for (int mi = 0; mi < 2; ++mi)
            #pragma unroll
            for (int ni = 0; ni < 4; ++ni)
                acc[mi][ni] = __builtin_amdgcn_wmma_f32_16x16x32_bf16(
                    false, af[mi], false, bfr[ni], (short)0, acc[mi][ni], false, false);
        __syncthreads();
    }

    int Mb = (lane >> 4) * 8;
    int nl = lane & 15;
    #pragma unroll
    for (int mi = 0; mi < 2; ++mi)
        #pragma unroll
        for (int ni = 0; ni < 4; ++ni)
            #pragma unroll
            for (int r = 0; r < 8; ++r) {
                int i = row0 + wm + mi * 16 + Mb + r;
                int d = d0 + wn + ni * 16 + nl;
                out[(size_t)i * DDIM + d] = acc[mi][ni][r];
            }
}

// ---------------- host-side launch ----------------

extern "C" void kernel_launch(void* const* d_in, const int* in_sizes, int n_in,
                              void* d_out, int out_size, void* d_ws, size_t ws_size,
                              hipStream_t stream) {
    const float* x = (const float*)d_in[0];
    const float* y = (const float*)d_in[1];

    char* ws = (char*)d_ws;
    size_t off = 0;
    __bf16* xbf = (__bf16*)(ws + off); off += (size_t)NPTS * DDIM * sizeof(__bf16);
    __bf16* ybf = (__bf16*)(ws + off); off += (size_t)NPTS * DDIM * sizeof(__bf16);
    float* nx     = (float*)(ws + off); off += (size_t)NPTS * sizeof(float);
    float* ny     = (float*)(ws + off); off += (size_t)NPTS * sizeof(float);
    float* rowLSE = (float*)(ws + off); off += (size_t)NPTS * sizeof(float);
    float* colLSE = (float*)(ws + off); off += (size_t)TWO_N * sizeof(float);
    off = (off + 255) & ~(size_t)255;
    float* L = (float*)(ws + off);      // 4096 x 8192 fp32 = 128 MB (fits in 192 MB L2)

    float* out = (float*)d_out;

    cvt_norm_kernel<<<dim3(NPTS, 2), 256, 0, stream>>>(x, y, xbf, ybf, nx, ny);
    logits_kernel<<<dim3(TWO_N / BN, NPTS / BM), 256, 0, stream>>>(xbf, ybf, nx, ny, L);
    row_lse_kernel<<<NPTS, 256, 0, stream>>>(L, rowLSE);
    col_lse_kernel<<<TWO_N / 256, 256, 0, stream>>>(L, colLSE);
    drift_kernel<<<dim3(DDIM / BN, NPTS / BM), 256, 0, stream>>>(xbf, ybf, L, rowLSE, colLSE, out);
}
